// MAP3d_57140244906023
// MI455X (gfx1250) — matrix operand order repaired and checked by reading, not compiled, not run on hardware
//
#include <hip/hip_runtime.h>

// ---------------- problem constants (fixed by reference) ----------------
#define BB   2
#define CC   64
#define HH   48
#define NN   (48*48*48)        // 110592 positions per batch
#define PP   (BB*NN)           // 221184 total positions
#define PC   (PP*CC)           // elements in a (B,N,C) f32 buffer
#define NC   (NN*CC)

// weight-tile table (prepacked bf16, WMMA B lane layout)
#define WT_P1  0               // 8 tiles: nt(4) x kt(2)
#define WT_CO  8               // 8 tiles
#define WT_P2  16              // 8 tiles
#define WT_F1  24              // 6 tiles: kt(6), nt=0
#define WT_NUM 30

// ---------------- WMMA plumbing (gfx1250, wave32) ----------------
typedef __attribute__((ext_vector_type(16))) __bf16          bf16x16;
typedef __attribute__((ext_vector_type(8)))  float           f32x8;

__device__ inline __bf16 cvt_bf16(float f) { return (__bf16)f; }   // native v_cvt path

__device__ inline f32x8 wmma_bf16(bf16x16 a, bf16x16 b, f32x8 c) {
    // v_wmma_f32_16x16x32_bf16  D = A(16x32) * B(32x16) + C
    return __builtin_amdgcn_wmma_f32_16x16x32_bf16(
        false, a, false, b, (short)0, c, false, false);
}

// A-matrix 16x32 bf16 (ISA 7.12.2): lane m=L&15, half=L>>4;
// VGPR v holds K pair: k = (v<4 ? 2v : 8+2v) + 8*half  (+k0)
__device__ inline bf16x16 make_A(const float* __restrict__ base, int stride,
                                 int k0, int lane) {
    int m = lane & 15, hf = lane >> 4;
    const float* row = base + m * stride + k0 + 8 * hf;
    bf16x16 r;
#pragma unroll
    for (int v = 0; v < 8; ++v) {
        int k = (v < 4 ? 2 * v : 8 + 2 * v);
        float2 f = *(const float2*)(row + k);
        r[2 * v]     = cvt_bf16(f.x);
        r[2 * v + 1] = cvt_bf16(f.y);
    }
    return r;
}

// prepacked B tile: ((tile*32 + lane)*16) bf16 elements, 32B aligned per lane
__device__ inline bf16x16 load_B(const unsigned short* __restrict__ pre,
                                 int tile, int lane) {
    return *(const bf16x16*)(pre + ((long)tile * 32 + lane) * 16);
}

__device__ inline float gelu_exact(float x) {
    return 0.5f * x * (1.0f + erff(x * 0.70710678118654752f));
}

// ---------------- kernel 0: prepack GEMM weights into bf16 WMMA-B layout ----------------
// one 32-thread block per tile; B[k][n] = W[n0+n][k0+k]
// lane: n = L&15, kh = (L>>4)*8 ; elem e -> k = kh + (e<8 ? e : e+8)
__global__ void k_prep_w(const float* __restrict__ p1w, const float* __restrict__ cow,
                         const float* __restrict__ p2w, const float* __restrict__ f1w,
                         unsigned short* __restrict__ pre) {
    int t = blockIdx.x, lane = threadIdx.x;
    const float* w; int Cin, n0, k0;
    if (t < 8)       { w = p1w; Cin = 64;  n0 = (t >> 1) * 16;        k0 = (t & 1) * 32; }
    else if (t < 16) { w = cow; Cin = 64;  n0 = ((t - 8) >> 1) * 16;  k0 = (t & 1) * 32; }
    else if (t < 24) { w = p2w; Cin = 64;  n0 = ((t - 16) >> 1) * 16; k0 = (t & 1) * 32; }
    else             { w = f1w; Cin = 192; n0 = 0;                    k0 = (t - 24) * 32; }
    int n = n0 + (lane & 15), kh = (lane >> 4) * 8;
    unsigned short* dst = pre + ((long)t * 32 + lane) * 16;
#pragma unroll
    for (int e = 0; e < 16; ++e)
        dst[e] = __builtin_bit_cast(unsigned short, cvt_bf16(w[n * Cin + k0 + kh + (e < 8 ? e : e + 8)]));
}

// ---------------- kernel 1: proj_1 GEMM + bias + GN partial sums ----------------
// block = 256 (8 waves), wave handles one 16-position tile; 4 N-tiles x 2 K-WMMAs
__global__ void k_gemm_p1(const float* __restrict__ x,
                          const unsigned short* __restrict__ pre,
                          const float* __restrict__ bias, float* __restrict__ y,
                          float* __restrict__ stats) {
    __shared__ float s_sum[8], s_sq[8];
    if (threadIdx.x < 8) { s_sum[threadIdx.x] = 0.f; s_sq[threadIdx.x] = 0.f; }
    __syncthreads();

    int lane = threadIdx.x & 31, wave = threadIdx.x >> 5;
    long pbase = ((long)blockIdx.x * 8 + wave) * 16;
    int b = (int)(pbase / NN);

    const float* asrc = x + pbase * CC;
    bf16x16 A0 = make_A(asrc, CC, 0, lane);
    bf16x16 A1 = make_A(asrc, CC, 32, lane);

    int n = lane & 15, hf = lane >> 4;
#pragma unroll
    for (int nt = 0; nt < 4; ++nt) {
        f32x8 acc = {};
        acc = wmma_bf16(A0, load_B(pre, WT_P1 + nt * 2 + 0, lane), acc);
        acc = wmma_bf16(A1, load_B(pre, WT_P1 + nt * 2 + 1, lane), acc);
        int c = nt * 16 + n;
        float bb = bias[c], s = 0.f, q = 0.f;
#pragma unroll
        for (int v = 0; v < 8; ++v) {
            float val = acc[v] + bb;
            int m = v + 8 * hf;
            y[(pbase + m) * CC + c] = val;
            s += val; q += val * val;
        }
        atomicAdd(&s_sum[c >> 3], s);   // ds_add_f32
        atomicAdd(&s_sq[c >> 3], q);
    }
    __syncthreads();
    if (threadIdx.x < 8) {
        atomicAdd(&stats[b * 16 + threadIdx.x],     s_sum[threadIdx.x]);
        atomicAdd(&stats[b * 16 + 8 + threadIdx.x], s_sq[threadIdx.x]);
    }
}

// ---------------- kernel 2: GroupNorm apply + exact GELU (in place) ----------------
__global__ void k_gn_gelu(float* __restrict__ y, const float* __restrict__ stats,
                          const float* __restrict__ g, const float* __restrict__ be) {
    long idx = (long)blockIdx.x * blockDim.x + threadIdx.x;
    if (idx >= (long)PC) return;
    int c = (int)(idx & 63);
    long p = idx >> 6;
    int b = (int)(p / NN), gr = c >> 3;
    const float cnt = 8.0f * (float)NN;
    float mu  = stats[b * 16 + gr] / cnt;
    float var = stats[b * 16 + 8 + gr] / cnt - mu * mu;
    float v = y[idx];
    float t = (v - mu) * rsqrtf(var + 1e-5f) * g[c] + be[c];
    y[idx] = gelu_exact(t);
}

// ---------------- kernel 3: fused depthwise 3/5/7 conv ----------------
__global__ void k_dwconv(const float* __restrict__ act,
                         const float* __restrict__ w3, const float* __restrict__ b3,
                         const float* __restrict__ w5, const float* __restrict__ b5,
                         const float* __restrict__ w7, const float* __restrict__ b7,
                         float* __restrict__ a3, float* __restrict__ a5,
                         float* __restrict__ a7) {
    long tid = (long)blockIdx.x * blockDim.x + threadIdx.x;
    if (tid >= (long)PC) return;
    int c = (int)(tid & 63);
    long p = tid >> 6;
    int b = (int)(p / NN);
    int nn = (int)(p % NN);
    int h = nn / (HH * HH), wq = (nn / HH) % HH, d = nn % HH;

    float s3 = 0.f, s5 = 0.f, s7 = 0.f;
    const long bbase = (long)b * NN;
#pragma unroll 1
    for (int i = 0; i < 7; ++i) {
        int hh = h + i - 3; if (hh < 0 || hh >= HH) continue;
#pragma unroll 1
        for (int j = 0; j < 7; ++j) {
            int ww = wq + j - 3; if (ww < 0 || ww >= HH) continue;
            long row = bbase + ((long)hh * HH + ww) * HH;
#pragma unroll 1
            for (int l = 0; l < 7; ++l) {
                int dd = d + l - 3; if (dd < 0 || dd >= HH) continue;
                float v = act[(row + dd) * CC + c];
                s7 += v * w7[c * 343 + i * 49 + j * 7 + l];
                if ((unsigned)(i - 1) <= 4u && (unsigned)(j - 1) <= 4u && (unsigned)(l - 1) <= 4u)
                    s5 += v * w5[c * 125 + (i - 1) * 25 + (j - 1) * 5 + (l - 1)];
                if ((unsigned)(i - 2) <= 2u && (unsigned)(j - 2) <= 2u && (unsigned)(l - 2) <= 2u)
                    s3 += v * w3[c * 27 + (i - 2) * 9 + (j - 2) * 3 + (l - 2)];
            }
        }
    }
    a3[tid] = s3 + b3[c];
    a5[tid] = s5 + b5[c];
    a7[tid] = s7 + b7[c];
}

// ---------------- kernel 4: positional encoding (trilinear 8^3 -> 48^3 + LN) ----------------
__global__ void k_posenc(const float* __restrict__ pe, const float* __restrict__ lg,
                         const float* __restrict__ lb, float* __restrict__ posb) {
    __shared__ float ssum[64], ssq[64];
    int c = threadIdx.x;
    int p = blockIdx.x;
    int h = p / (HH * HH), wq = (p / HH) % HH, d = p % HH;

    int i0[3], i1[3]; float t[3]; int coord[3] = {h, wq, d};
#pragma unroll
    for (int ax = 0; ax < 3; ++ax) {
        float s = ((float)coord[ax] + 0.5f) * (8.0f / 48.0f) - 0.5f;
        float fl = floorf(s); t[ax] = s - fl;
        int a = (int)fl, b2 = a + 1;
        i0[ax] = min(max(a, 0), 7);
        i1[ax] = min(max(b2, 0), 7);
    }
    const float* base = pe + c * 512;
    float v000 = base[i0[0]*64 + i0[1]*8 + i0[2]], v001 = base[i0[0]*64 + i0[1]*8 + i1[2]];
    float v010 = base[i0[0]*64 + i1[1]*8 + i0[2]], v011 = base[i0[0]*64 + i1[1]*8 + i1[2]];
    float v100 = base[i1[0]*64 + i0[1]*8 + i0[2]], v101 = base[i1[0]*64 + i0[1]*8 + i1[2]];
    float v110 = base[i1[0]*64 + i1[1]*8 + i0[2]], v111 = base[i1[0]*64 + i1[1]*8 + i1[2]];
    float c00 = v000 + t[2] * (v001 - v000), c01 = v010 + t[2] * (v011 - v010);
    float c10 = v100 + t[2] * (v101 - v100), c11 = v110 + t[2] * (v111 - v110);
    float c0 = c00 + t[1] * (c01 - c00), c1 = c10 + t[1] * (c11 - c10);
    float val = c0 + t[0] * (c1 - c0);

    ssum[c] = val; ssq[c] = val * val;
    __syncthreads();
    for (int o = 32; o > 0; o >>= 1) {
        if (c < o) { ssum[c] += ssum[c + o]; ssq[c] += ssq[c + o]; }
        __syncthreads();
    }
    float mu = ssum[0] * (1.0f / 64.0f);
    float var = ssq[0] * (1.0f / 64.0f) - mu * mu;
    posb[(long)p * CC + c] = (val - mu) * rsqrtf(var + 1e-5f) * lg[c] + lb[c];
}

// ---------------- kernel 5: gating (f1 -> gelu -> f2 -> softmax -> blend -> co -> *identity) ----------------
// one wave per 16-position tile
__global__ void k_gating(const float* __restrict__ a3, const float* __restrict__ a5,
                         const float* __restrict__ a7, const float* __restrict__ act,
                         const unsigned short* __restrict__ pre,
                         const float* __restrict__ f1b,
                         const float* __restrict__ f2w, const float* __restrict__ f2b,
                         const float* __restrict__ cob,
                         float* __restrict__ gated) {
    __shared__ float lds_h[16][17];
    __shared__ float lds_w[16][4];
    int lane = threadIdx.x;
    long pbase = (long)blockIdx.x * 16;
    int n = lane & 15, hf = lane >> 4;

    // ---- f1: (16 pos x 192) x (192 x 16) via 6 WMMAs ----
    f32x8 acc = {};
    const float* srcs[3] = { a3, a5, a7 };
#pragma unroll
    for (int part = 0; part < 3; ++part) {
        const float* s = srcs[part] + pbase * CC;
        bf16x16 A0 = make_A(s, CC, 0, lane);
        bf16x16 A1 = make_A(s, CC, 32, lane);
        acc = wmma_bf16(A0, load_B(pre, WT_F1 + part * 2 + 0, lane), acc);
        acc = wmma_bf16(A1, load_B(pre, WT_F1 + part * 2 + 1, lane), acc);
    }
    float fb = f1b[n];
#pragma unroll
    for (int v = 0; v < 8; ++v)
        lds_h[v + 8 * hf][n] = gelu_exact(acc[v] + fb);
    __syncthreads();

    // ---- f2 (16 -> 3) + softmax, lanes 0..15 each handle one position ----
    if (lane < 16) {
        float s0 = f2b[0], s1 = f2b[1], s2 = f2b[2];
#pragma unroll
        for (int q = 0; q < 16; ++q) {
            float hq = lds_h[lane][q];
            s0 += f2w[q] * hq; s1 += f2w[16 + q] * hq; s2 += f2w[32 + q] * hq;
        }
        float mx = fmaxf(s0, fmaxf(s1, s2));
        float e0 = expf(s0 - mx), e1 = expf(s1 - mx), e2 = expf(s2 - mx);
        float inv = 1.0f / (e0 + e1 + e2);
        lds_w[lane][0] = e0 * inv; lds_w[lane][1] = e1 * inv; lds_w[lane][2] = e2 * inv;
    }
    __syncthreads();

    // ---- blend branches into bf16 A tiles of co GEMM ----
    int m = lane & 15;
    float w0 = lds_w[m][0], w1 = lds_w[m][1], w2 = lds_w[m][2];
    const float* p3 = a3 + (pbase + m) * CC + 8 * hf;
    const float* p5 = a5 + (pbase + m) * CC + 8 * hf;
    const float* p7 = a7 + (pbase + m) * CC + 8 * hf;
    bf16x16 A0, A1;
#pragma unroll
    for (int v = 0; v < 8; ++v) {
        int k = (v < 4 ? 2 * v : 8 + 2 * v);
        float2 x3 = *(const float2*)(p3 + k);
        float2 x5 = *(const float2*)(p5 + k);
        float2 x7 = *(const float2*)(p7 + k);
        A0[2 * v]     = cvt_bf16(w0 * x3.x + w1 * x5.x + w2 * x7.x);
        A0[2 * v + 1] = cvt_bf16(w0 * x3.y + w1 * x5.y + w2 * x7.y);
        x3 = *(const float2*)(p3 + k + 32);
        x5 = *(const float2*)(p5 + k + 32);
        x7 = *(const float2*)(p7 + k + 32);
        A1[2 * v]     = cvt_bf16(w0 * x3.x + w1 * x5.x + w2 * x7.x);
        A1[2 * v + 1] = cvt_bf16(w0 * x3.y + w1 * x5.y + w2 * x7.y);
    }

    // ---- co: 64x64 GEMM + bias, multiply by identity (act) ----
#pragma unroll
    for (int nt = 0; nt < 4; ++nt) {
        f32x8 acc2 = {};
        acc2 = wmma_bf16(A0, load_B(pre, WT_CO + nt * 2 + 0, lane), acc2);
        acc2 = wmma_bf16(A1, load_B(pre, WT_CO + nt * 2 + 1, lane), acc2);
        int c = nt * 16 + n;
        float cb = cob[c];
#pragma unroll
        for (int v = 0; v < 8; ++v) {
            int mm = v + 8 * hf;
            long idx = (pbase + mm) * CC + c;
            gated[idx] = (acc2[v] + cb) * act[idx];   // gated aliases a3: tile fully read above
        }
    }
}

// ---------------- kernel 6: proj_2 GEMM + pos add + bias + residual ----------------
__global__ void k_gemm_p2(const float* __restrict__ gated, const float* __restrict__ posb,
                          const unsigned short* __restrict__ pre,
                          const float* __restrict__ bias,
                          const float* __restrict__ xin, float* __restrict__ out) {
    int lane = threadIdx.x & 31, wave = threadIdx.x >> 5;
    long pbase = ((long)blockIdx.x * 8 + wave) * 16;
    int nmod = (int)(pbase % NN);
    int m = lane & 15, hf = lane >> 4, n = lane & 15;

    const float* gsrc = gated + (pbase + m) * CC + 8 * hf;
    const float* psrc = posb + ((long)(nmod + m)) * CC + 8 * hf;
    bf16x16 A0, A1;
#pragma unroll
    for (int v = 0; v < 8; ++v) {
        int k = (v < 4 ? 2 * v : 8 + 2 * v);
        float2 g0 = *(const float2*)(gsrc + k);
        float2 q0 = *(const float2*)(psrc + k);
        A0[2 * v]     = cvt_bf16(g0.x + q0.x);
        A0[2 * v + 1] = cvt_bf16(g0.y + q0.y);
        float2 g1 = *(const float2*)(gsrc + k + 32);
        float2 q1 = *(const float2*)(psrc + k + 32);
        A1[2 * v]     = cvt_bf16(g1.x + q1.x);
        A1[2 * v + 1] = cvt_bf16(g1.y + q1.y);
    }

#pragma unroll
    for (int nt = 0; nt < 4; ++nt) {
        f32x8 acc = {};
        acc = wmma_bf16(A0, load_B(pre, WT_P2 + nt * 2 + 0, lane), acc);
        acc = wmma_bf16(A1, load_B(pre, WT_P2 + nt * 2 + 1, lane), acc);
        int c = nt * 16 + n;
        float bb = bias[c];
#pragma unroll
        for (int v = 0; v < 8; ++v) {
            int mm = v + 8 * hf;
            long idx = (pbase + mm) * CC + c;
            out[idx] = acc[v] + bb + xin[idx];
        }
    }
}

// ---------------- host launcher ----------------
extern "C" void kernel_launch(void* const* d_in, const int* in_sizes, int n_in,
                              void* d_out, int out_size, void* d_ws, size_t ws_size,
                              hipStream_t stream) {
    const float* x     = (const float*)d_in[0];
    const float* p1_w  = (const float*)d_in[1];
    const float* p1_b  = (const float*)d_in[2];
    const float* gn_g  = (const float*)d_in[3];
    const float* gn_b  = (const float*)d_in[4];
    const float* dw3_w = (const float*)d_in[5];
    const float* dw3_b = (const float*)d_in[6];
    const float* dw5_w = (const float*)d_in[7];
    const float* dw5_b = (const float*)d_in[8];
    const float* dw7_w = (const float*)d_in[9];
    const float* dw7_b = (const float*)d_in[10];
    const float* f1_w  = (const float*)d_in[11];
    const float* f1_b  = (const float*)d_in[12];
    const float* f2_w  = (const float*)d_in[13];
    const float* f2_b  = (const float*)d_in[14];
    const float* co_w  = (const float*)d_in[15];
    const float* co_b  = (const float*)d_in[16];
    const float* pose  = (const float*)d_in[17];
    const float* ln_g  = (const float*)d_in[18];
    const float* ln_b  = (const float*)d_in[19];
    const float* p2_w  = (const float*)d_in[20];
    const float* p2_b  = (const float*)d_in[21];
    float* out = (float*)d_out;

    // workspace layout (f32): act | a3 | a5 | a7 | posb | stats(32) | wpre(bf16 tiles)
    float* act   = (float*)d_ws;
    float* a3    = act  + (long)PC;
    float* a5    = a3   + (long)PC;
    float* a7    = a5   + (long)PC;
    float* posb  = a7   + (long)PC;
    float* stats = posb + (long)NC;
    unsigned short* wpre = (unsigned short*)(stats + 32);
    float* gated = a3;   // alias: gating reads its tile before storing

    hipMemsetAsync(stats, 0, 32 * sizeof(float), stream);

    // prepack GEMM weights into bf16 WMMA-B lane layout (30 tiles, ~30KB)
    k_prep_w<<<WT_NUM, 32, 0, stream>>>(p1_w, co_w, p2_w, f1_w, wpre);

    // proj_1 GEMM + GN stats : P/16 tiles, 8 waves per 256-thread block
    k_gemm_p1<<<PP / 128, 256, 0, stream>>>(x, wpre, p1_b, act, stats);
    // GN apply + GELU
    k_gn_gelu<<<(PC + 255) / 256, 256, 0, stream>>>(act, stats, gn_g, gn_b);
    // fused depthwise convs
    k_dwconv<<<(PC + 255) / 256, 256, 0, stream>>>(act, dw3_w, dw3_b, dw5_w, dw5_b,
                                                   dw7_w, dw7_b, a3, a5, a7);
    // positional encoding (independent)
    k_posenc<<<NN, 64, 0, stream>>>(pose, ln_g, ln_b, posb);
    // gating: one wave per 16-position tile
    k_gating<<<PP / 16, 32, 0, stream>>>(a3, a5, a7, act, wpre, f1_b, f2_w, f2_b,
                                         co_b, gated);
    // proj_2 + residual -> d_out
    k_gemm_p2<<<PP / 128, 256, 0, stream>>>(gated, posb, wpre, p2_b, x, out);
}
